// TransformerEncoderLayer_2525440770595
// MI455X (gfx1250) — matrix operand order repaired and checked by reading, not compiled, not run on hardware
//
#include <hip/hip_runtime.h>
#include <hip/hip_bf16.h>
#include <stdint.h>

typedef __bf16 bf16;
typedef __attribute__((ext_vector_type(16))) __bf16 bf16x16;
typedef __attribute__((ext_vector_type(8)))  float  floatx8;
typedef __attribute__((ext_vector_type(4)))  unsigned int u32x4;

union BF16Frag { bf16x16 v; u32x4 q[2]; };

__device__ __forceinline__ bf16 f2bf(float f) {
    unsigned u = __float_as_uint(f);
    unsigned r = u + 0x7FFFu + ((u >> 16) & 1u);   // round-to-nearest-even
    unsigned short h = (unsigned short)(r >> 16);
    return __builtin_bit_cast(bf16, h);
}

// XOR lane swizzle within 16-lane groups via ds_swizzle_b32 (group-of-32 mode:
// offset = xor_mask<<10 | and_mask 0x1f). One DS-pipe op, no index VGPR.
template<int IMM>
__device__ __forceinline__ float swz_xor(float v) {
    return __int_as_float(__builtin_amdgcn_ds_swizzle(__float_as_int(v), IMM));
}
#define SWZ_X1 0x041F
#define SWZ_X2 0x081F
#define SWZ_X4 0x101F
#define SWZ_X8 0x201F

// ---------------------------------------------------------------------------
// CDNA5 async global->LDS copy (ASYNCcnt-tracked, no VGPR round trip).
// ---------------------------------------------------------------------------
__device__ __forceinline__ void async_b128(const bf16* gptr, bf16* lptr) {
    unsigned lds = (unsigned)(uintptr_t)lptr;
    unsigned long long ga = (unsigned long long)(uintptr_t)gptr;
    asm volatile("global_load_async_to_lds_b128 %0, %1, off"
                 :: "v"(lds), "v"(ga) : "memory");
}
__device__ __forceinline__ void wait_async0() {
    asm volatile("s_wait_asynccnt 0x0" ::: "memory");
}

// ---------------------------------------------------------------------------
// fp32 -> bf16 conversion (grid-stride, for activations)
// ---------------------------------------------------------------------------
__global__ void cvt_bf16_kernel(const float* __restrict__ src, bf16* __restrict__ dst, int n) {
    int i = blockIdx.x * blockDim.x + threadIdx.x;
    int stride = gridDim.x * blockDim.x;
    for (; i < n; i += stride) dst[i] = f2bf(src[i]);
}

// ---------------------------------------------------------------------------
// fp32 [K,N] -> bf16 transposed [N,K] (weights; once per call). 32x32 tiles.
// ---------------------------------------------------------------------------
__global__ __launch_bounds__(256) void transpose_cvt_kernel(
    const float* __restrict__ W, bf16* __restrict__ Wt, int K, int N)
{
    __shared__ float tile[32][33];
    const int k0 = blockIdx.y * 32, n0 = blockIdx.x * 32;
    const int r = threadIdx.x >> 3, c0 = (threadIdx.x & 7) * 4;
    const float* src = W + (size_t)(k0 + r) * N + n0 + c0;
    #pragma unroll
    for (int i = 0; i < 4; ++i) tile[r][c0 + i] = src[i];
    __syncthreads();
    bf16* dst = Wt + (size_t)(n0 + r) * K + k0 + c0;
    #pragma unroll
    for (int i = 0; i < 4; ++i) dst[i] = f2bf(tile[c0 + i][r]);
}

// ---------------------------------------------------------------------------
// WMMA GEMM: C[M,N] = A[M,K] * Bt[N,K]^T + bias. 128x128 block tile, K-step 32,
// 8 waves of 32(M)x64(N). Double-buffered LDS fed by async b128 copies; one
// barrier per K-step. Epilogue flags are template params (branch-free).
// ---------------------------------------------------------------------------
#define LDA 40   // 32 + 8 pad; row stride 80B (16B-aligned)

template<int RELU, int STF, int STB, int STBT>
__global__ __launch_bounds__(256) void gemm_bf16_wmma(
    const bf16* __restrict__ A, const bf16* __restrict__ Bt,
    const float* __restrict__ bias,
    float* __restrict__ Cf, bf16* __restrict__ Cb,
    int M, int N, int K)
{
    __shared__ __align__(16) bf16 As[2][128 * LDA];
    __shared__ __align__(16) bf16 Bs[2][128 * LDA];

    const int tid  = threadIdx.x;
    const int lane = tid & 31;
    const int wid  = tid >> 5;
    const int half = lane >> 4;
    const int ln   = lane & 15;

    const int m0 = blockIdx.y * 128;
    const int n0 = blockIdx.x * 128;
    const int wm = (wid >> 1) * 32;
    const int wn = (wid & 1) * 64;

    floatx8 acc[2][4];
    for (int mt = 0; mt < 2; ++mt)
        for (int nt = 0; nt < 4; ++nt)
            for (int r = 0; r < 8; ++r) acc[mt][nt][r] = 0.0f;

    const int srow  = tid >> 1;
    const int shalf = (tid & 1) * 16;

    auto stage = [&](int buf, int k0) {
        const bf16* ap = A  + (size_t)(m0 + srow) * K + k0 + shalf;
        bf16* ad = &As[buf][srow * LDA + shalf];
        async_b128(ap,     ad);
        async_b128(ap + 8, ad + 8);
        const bf16* bp = Bt + (size_t)(n0 + srow) * K + k0 + shalf;
        bf16* bd = &Bs[buf][srow * LDA + shalf];
        async_b128(bp,     bd);
        async_b128(bp + 8, bd + 8);
    };

    stage(0, 0);
    for (int k0 = 0; k0 < K; k0 += 32) {
        const int cur = (k0 >> 5) & 1;
        wait_async0();
        __syncthreads();
        if (k0 + 32 < K) stage(cur ^ 1, k0 + 32);

        BF16Frag af[2], bfr[4];
        #pragma unroll
        for (int mt = 0; mt < 2; ++mt) {
            const bf16* p = &As[cur][(wm + mt * 16 + ln) * LDA];
            af[mt].q[0] = *(const u32x4*)(p + half * 16);
            af[mt].q[1] = *(const u32x4*)(p + half * 16 + 8);
        }
        #pragma unroll
        for (int nt = 0; nt < 4; ++nt) {
            const bf16* p = &Bs[cur][(wn + nt * 16 + ln) * LDA + half * 16];
            bfr[nt].q[0] = *(const u32x4*)(p);
            bfr[nt].q[1] = *(const u32x4*)(p + 8);
        }
        #pragma unroll
        for (int mt = 0; mt < 2; ++mt)
            #pragma unroll
            for (int nt = 0; nt < 4; ++nt)
                acc[mt][nt] = __builtin_amdgcn_wmma_f32_16x16x32_bf16(
                    false, af[mt].v, false, bfr[nt].v, (short)0, acc[mt][nt], false, false);
    }

    float bsv[4];
    #pragma unroll
    for (int nt = 0; nt < 4; ++nt)
        bsv[nt] = bias[n0 + wn + nt * 16 + ln];

    for (int mt = 0; mt < 2; ++mt)
        for (int nt = 0; nt < 4; ++nt)
            #pragma unroll
            for (int r = 0; r < 8; ++r) {
                int gr = m0 + wm + mt * 16 + half * 8 + r;
                int gc = n0 + wn + nt * 16 + ln;
                float v = acc[mt][nt][r] + bsv[nt];
                if (RELU) v = fmaxf(v, 0.0f);
                if (STF)  Cf[(size_t)gr * N + gc] = v;
                if (STB)  Cb[(size_t)gr * N + gc] = f2bf(v);
                if (STBT) Cb[(size_t)gc * M + gr] = f2bf(v);
            }
}

// ---------------------------------------------------------------------------
// Flash attention: one (b, h, 128-query tile) per block, 8 waves x 16 q-rows.
// Double-buffered 32-key K/V tiles via async b128 (V pre-transposed). Online
// softmax with ds_swizzle xor-reductions in 16-lane groups.
// ---------------------------------------------------------------------------
#define SEQ 2048
#define DM  1024
#define MR  4096

__global__ __launch_bounds__(256) void attn_flash_wmma(
    const bf16* __restrict__ Q, const bf16* __restrict__ Kmat,
    const bf16* __restrict__ VT,        // [DM][MR] = V^T
    const int* __restrict__ mask, bf16* __restrict__ ctx)
{
    __shared__ __align__(16) bf16 Ks [2][32 * 72];
    __shared__ __align__(16) bf16 Vt [2][64 * 40];
    __shared__ __align__(16) bf16 Pst[8 * 16 * 40];

    const int tid  = threadIdx.x;
    const int lane = tid & 31;
    const int wid  = tid >> 5;
    const int half = lane >> 4;
    const int ln   = lane & 15;

    const int h  = blockIdx.y;
    const int b  = blockIdx.z;
    const int q0 = blockIdx.x * 128 + wid * 16;

    const size_t rowQ = ((size_t)b * SEQ + q0 + ln) * DM + (size_t)h * 64;
    BF16Frag qf[2];
    #pragma unroll
    for (int j = 0; j < 2; ++j) {
        const bf16* p = Q + rowQ + j * 32;
        qf[j].q[0] = *(const u32x4*)(p + half * 16);
        qf[j].q[1] = *(const u32x4*)(p + half * 16 + 8);
    }

    floatx8 o[4];
    for (int dt = 0; dt < 4; ++dt)
        for (int r = 0; r < 8; ++r) o[dt][r] = 0.0f;
    float mrow[8], lrow[8];
    for (int r = 0; r < 8; ++r) { mrow[r] = -3.0e38f; lrow[r] = 0.0f; }

    const int sk = tid >> 3, scg = (tid & 7) * 8;
    const int vd = tid >> 2, vq  = (tid & 3) * 8;

    auto stage = [&](int buf, int kt) {
        const bf16* kp = Kmat + ((size_t)b * SEQ + kt + sk) * DM + (size_t)h * 64 + scg;
        async_b128(kp, &Ks[buf][sk * 72 + scg]);
        const bf16* vp = VT + ((size_t)h * 64 + vd) * MR + (size_t)b * SEQ + kt + vq;
        async_b128(vp, &Vt[buf][vd * 40 + vq]);
    };

    stage(0, 0);
    for (int kt = 0; kt < SEQ; kt += 32) {
        const int cur = (kt >> 5) & 1;
        wait_async0();
        __syncthreads();
        if (kt + 32 < SEQ) stage(cur ^ 1, kt + 32);

        // hoist all four K fragments, then issue the four score WMMAs
        BF16Frag kf[2][2];
        #pragma unroll
        for (int j = 0; j < 2; ++j)
            #pragma unroll
            for (int nt = 0; nt < 2; ++nt) {
                const bf16* p = &Ks[cur][(nt * 16 + ln) * 72 + j * 32 + half * 16];
                kf[j][nt].q[0] = *(const u32x4*)(p);
                kf[j][nt].q[1] = *(const u32x4*)(p + 8);
            }
        floatx8 s0, s1;
        for (int r = 0; r < 8; ++r) { s0[r] = 0.0f; s1[r] = 0.0f; }
        s0 = __builtin_amdgcn_wmma_f32_16x16x32_bf16(false, qf[0].v, false, kf[0][0].v, (short)0, s0, false, false);
        s1 = __builtin_amdgcn_wmma_f32_16x16x32_bf16(false, qf[0].v, false, kf[0][1].v, (short)0, s1, false, false);
        s0 = __builtin_amdgcn_wmma_f32_16x16x32_bf16(false, qf[1].v, false, kf[1][0].v, (short)0, s0, false, false);
        s1 = __builtin_amdgcn_wmma_f32_16x16x32_bf16(false, qf[1].v, false, kf[1][1].v, (short)0, s1, false, false);

        // mask as additive bias -> scale+mask folds into one fma per element
        const float mb0 = (mask[b * SEQ + kt + ln]      == 0) ? -1.0e9f : 0.0f;
        const float mb1 = (mask[b * SEQ + kt + 16 + ln] == 0) ? -1.0e9f : 0.0f;

        #pragma unroll
        for (int r = 0; r < 8; ++r) {
            float a0 = fmaf(s0[r], 0.125f, mb0);
            float a1 = fmaf(s1[r], 0.125f, mb1);
            float tm = fmaxf(a0, a1);
            tm = fmaxf(tm, swz_xor<SWZ_X1>(tm));
            tm = fmaxf(tm, swz_xor<SWZ_X2>(tm));
            tm = fmaxf(tm, swz_xor<SWZ_X4>(tm));
            tm = fmaxf(tm, swz_xor<SWZ_X8>(tm));
            float mn = fmaxf(mrow[r], tm);
            float p0 = __expf(a0 - mn);
            float p1 = __expf(a1 - mn);
            float rs = p0 + p1;
            rs += swz_xor<SWZ_X1>(rs);
            rs += swz_xor<SWZ_X2>(rs);
            rs += swz_xor<SWZ_X4>(rs);
            rs += swz_xor<SWZ_X8>(rs);
            float alpha = __expf(mrow[r] - mn);
            lrow[r] = lrow[r] * alpha + rs;
            mrow[r] = mn;
            #pragma unroll
            for (int dt = 0; dt < 4; ++dt) o[dt][r] *= alpha;
            int prow = half * 8 + r;
            Pst[wid * 640 + prow * 40 + ln]      = f2bf(p0);
            Pst[wid * 640 + prow * 40 + 16 + ln] = f2bf(p1);
        }
        __builtin_amdgcn_wave_barrier();   // keep DS write->read order in-wave

        BF16Frag pf;
        {
            const bf16* p = &Pst[wid * 640 + ln * 40];
            pf.q[0] = *(const u32x4*)(p + half * 16);
            pf.q[1] = *(const u32x4*)(p + half * 16 + 8);
        }
        // hoist all four V fragments, then the four P*V WMMAs
        BF16Frag vf[4];
        #pragma unroll
        for (int dt = 0; dt < 4; ++dt) {
            const bf16* p = &Vt[cur][(dt * 16 + ln) * 40 + half * 16];
            vf[dt].q[0] = *(const u32x4*)(p);
            vf[dt].q[1] = *(const u32x4*)(p + 8);
        }
        #pragma unroll
        for (int dt = 0; dt < 4; ++dt)
            o[dt] = __builtin_amdgcn_wmma_f32_16x16x32_bf16(false, pf.v, false, vf[dt].v, (short)0, o[dt], false, false);
    }

    float invl[8];
    #pragma unroll
    for (int r = 0; r < 8; ++r) invl[r] = 1.0f / lrow[r];
    #pragma unroll
    for (int dt = 0; dt < 4; ++dt)
        for (int r = 0; r < 8; ++r) {
            int row = q0 + half * 8 + r;
            int col = h * 64 + dt * 16 + ln;
            ctx[((size_t)b * SEQ + row) * DM + col] = f2bf(o[dt][r] * invl[r]);
        }
}

// ---------------------------------------------------------------------------
// y = LayerNorm(X + R) * g + be ; f32 out (+ optional bf16 copy)
// ---------------------------------------------------------------------------
template<int STB>
__global__ __launch_bounds__(256) void residual_ln_kernel(
    const float* __restrict__ X, const float* __restrict__ R,
    const float* __restrict__ g, const float* __restrict__ be,
    float* __restrict__ outF, bf16* __restrict__ outB)
{
    __shared__ float red[256];
    const int tid = threadIdx.x;
    const size_t base = (size_t)blockIdx.x * 1024;
    float v[4];
    float s = 0.0f;
    #pragma unroll
    for (int i = 0; i < 4; ++i) {
        int c = tid + i * 256;
        v[i] = X[base + c] + R[base + c];
        s += v[i];
    }
    red[tid] = s; __syncthreads();
    for (int off = 128; off > 0; off >>= 1) {
        if (tid < off) red[tid] += red[tid + off];
        __syncthreads();
    }
    float mu = red[0] * (1.0f / 1024.0f);
    __syncthreads();
    float s2 = 0.0f;
    #pragma unroll
    for (int i = 0; i < 4; ++i) { float d = v[i] - mu; s2 += d * d; }
    red[tid] = s2; __syncthreads();
    for (int off = 128; off > 0; off >>= 1) {
        if (tid < off) red[tid] += red[tid + off];
        __syncthreads();
    }
    float inv = rsqrtf(red[0] * (1.0f / 1024.0f) + 1e-5f);
    #pragma unroll
    for (int i = 0; i < 4; ++i) {
        int c = tid + i * 256;
        float y = (v[i] - mu) * inv * g[c] + be[c];
        outF[base + c] = y;
        if (STB) outB[base + c] = f2bf(y);
    }
}

// ---------------------------------------------------------------------------
// Orchestration
// ---------------------------------------------------------------------------
extern "C" void kernel_launch(void* const* d_in, const int* in_sizes, int n_in,
                              void* d_out, int out_size, void* d_ws, size_t ws_size,
                              hipStream_t stream) {
    (void)in_sizes; (void)n_in; (void)out_size; (void)ws_size;
    const float* x   = (const float*)d_in[0];
    const int*   msk = (const int*)  d_in[1];
    const float* Wq  = (const float*)d_in[2];
    const float* bq  = (const float*)d_in[3];
    const float* Wk  = (const float*)d_in[4];
    const float* bk  = (const float*)d_in[5];
    const float* Wv  = (const float*)d_in[6];
    const float* bv  = (const float*)d_in[7];
    const float* Wo  = (const float*)d_in[8];
    const float* bo  = (const float*)d_in[9];
    const float* W1  = (const float*)d_in[10];
    const float* b1  = (const float*)d_in[11];
    const float* W2  = (const float*)d_in[12];
    const float* b2  = (const float*)d_in[13];
    const float* g1  = (const float*)d_in[14];
    const float* be1 = (const float*)d_in[15];
    const float* g2  = (const float*)d_in[16];
    const float* be2 = (const float*)d_in[17];
    float* out = (float*)d_out;

    const int Mr = 4096, D = 1024, F = 4096;

    char* w = (char*)d_ws;
    auto alloc = [&](size_t bytes) { void* p = (void*)w; w += (bytes + 255) & ~(size_t)255; return p; };
    bf16*  xb   = (bf16*) alloc((size_t)Mr * D * 2);
    bf16*  wqT  = (bf16*) alloc((size_t)D * D * 2);
    bf16*  wkT  = (bf16*) alloc((size_t)D * D * 2);
    bf16*  wvT  = (bf16*) alloc((size_t)D * D * 2);
    bf16*  woT  = (bf16*) alloc((size_t)D * D * 2);
    bf16*  w1T  = (bf16*) alloc((size_t)D * F * 2);   // [F][D]
    bf16*  w2T  = (bf16*) alloc((size_t)F * D * 2);   // [D][F]
    bf16*  qb   = (bf16*) alloc((size_t)Mr * D * 2);
    bf16*  kb   = (bf16*) alloc((size_t)Mr * D * 2);
    bf16*  vbT  = (bf16*) alloc((size_t)Mr * D * 2);  // [D][Mr] = V^T
    bf16*  ctxb = (bf16*) alloc((size_t)Mr * D * 2);
    float* aof  = (float*)alloc((size_t)Mr * D * 4);
    float* hf   = (float*)alloc((size_t)Mr * D * 4);
    bf16*  hb   = (bf16*) alloc((size_t)Mr * D * 2);
    bf16*  ffb  = (bf16*) alloc((size_t)Mr * F * 2);
    float* f2f  = aof;

    dim3 blk(256);
    cvt_bf16_kernel<<<dim3(4096), blk, 0, stream>>>(x, xb, Mr * D);
    transpose_cvt_kernel<<<dim3(D / 32, D / 32), blk, 0, stream>>>(Wq, wqT, D, D);
    transpose_cvt_kernel<<<dim3(D / 32, D / 32), blk, 0, stream>>>(Wk, wkT, D, D);
    transpose_cvt_kernel<<<dim3(D / 32, D / 32), blk, 0, stream>>>(Wv, wvT, D, D);
    transpose_cvt_kernel<<<dim3(D / 32, D / 32), blk, 0, stream>>>(Wo, woT, D, D);
    transpose_cvt_kernel<<<dim3(F / 32, D / 32), blk, 0, stream>>>(W1, w1T, D, F);
    transpose_cvt_kernel<<<dim3(D / 32, F / 32), blk, 0, stream>>>(W2, w2T, F, D);

    dim3 gP(D / 128, Mr / 128);
    gemm_bf16_wmma<0,0,1,0><<<gP, blk, 0, stream>>>(xb, wqT, bq, nullptr, qb,  Mr, D, D);
    gemm_bf16_wmma<0,0,1,0><<<gP, blk, 0, stream>>>(xb, wkT, bk, nullptr, kb,  Mr, D, D);
    gemm_bf16_wmma<0,0,0,1><<<gP, blk, 0, stream>>>(xb, wvT, bv, nullptr, vbT, Mr, D, D);

    dim3 gA(SEQ / 128, 16, 2);
    attn_flash_wmma<<<gA, blk, 0, stream>>>(qb, kb, vbT, msk, ctxb);

    gemm_bf16_wmma<0,1,0,0><<<gP, blk, 0, stream>>>(ctxb, woT, bo, aof, nullptr, Mr, D, D);
    residual_ln_kernel<1><<<dim3(Mr), blk, 0, stream>>>(x, aof, g1, be1, hf, hb);

    dim3 gF1(F / 128, Mr / 128);
    gemm_bf16_wmma<1,0,1,0><<<gF1, blk, 0, stream>>>(hb, w1T, b1, nullptr, ffb, Mr, F, D);
    gemm_bf16_wmma<0,1,0,0><<<gP,  blk, 0, stream>>>(ffb, w2T, b2, f2f, nullptr, Mr, D, F);
    residual_ln_kernel<0><<<dim3(Mr), blk, 0, stream>>>(hf, f2f, g2, be2, out, nullptr);
}
